// MultiAttnLayer_6236292514545
// MI455X (gfx1250) — compile-verified, matrix operand order checked by reading, not run on hardware
//
#include <hip/hip_runtime.h>
#include <hip/hip_bf16.h>

typedef __attribute__((ext_vector_type(16))) __bf16 v16bf;
typedef __attribute__((ext_vector_type(8)))  __bf16 v8bf;
typedef __attribute__((ext_vector_type(8)))  float  v8f;

union BFrag {
    v16bf v;
    v8bf  h[2];
};

#define LOG2E 1.4426950408889634f

// Async 16B global -> LDS copy (ASYNCcnt path, no VGPR round trip).
__device__ __forceinline__ void async_b128(const void* src, void* lds_dst) {
    unsigned dst = (unsigned)(size_t)lds_dst;   // low 32 bits = LDS offset
    asm volatile("global_load_async_to_lds_b128 %0, %1, off"
                 :: "v"(dst), "v"(src) : "memory");
}
__device__ __forceinline__ void wait_async0() {
    asm volatile("s_wait_asynccnt 0" ::: "memory");
}

// DPP butterfly reductions across each 16-lane row (VALU-only, no LDS).
// Codes: 0xB1 = quad_perm(1,0,3,2), 0x4E = quad_perm(2,3,0,1),
//        0x141 = row_half_mirror,   0x140 = row_mirror.
__device__ __forceinline__ float dpp_max16(float v) {
    v = fmaxf(v, __int_as_float(__builtin_amdgcn_update_dpp(0, __float_as_int(v), 0xB1,  0xF, 0xF, true)));
    v = fmaxf(v, __int_as_float(__builtin_amdgcn_update_dpp(0, __float_as_int(v), 0x4E,  0xF, 0xF, true)));
    v = fmaxf(v, __int_as_float(__builtin_amdgcn_update_dpp(0, __float_as_int(v), 0x141, 0xF, 0xF, true)));
    v = fmaxf(v, __int_as_float(__builtin_amdgcn_update_dpp(0, __float_as_int(v), 0x140, 0xF, 0xF, true)));
    return v;
}
__device__ __forceinline__ float dpp_sum16(float v) {
    v += __int_as_float(__builtin_amdgcn_update_dpp(0, __float_as_int(v), 0xB1,  0xF, 0xF, true));
    v += __int_as_float(__builtin_amdgcn_update_dpp(0, __float_as_int(v), 0x4E,  0xF, 0xF, true));
    v += __int_as_float(__builtin_amdgcn_update_dpp(0, __float_as_int(v), 0x141, 0xF, 0xF, true));
    v += __int_as_float(__builtin_amdgcn_update_dpp(0, __float_as_int(v), 0x140, 0xF, 0xF, true));
    return v;
}

// =====================================================================
// One-time weight transpose+convert: Wt[n][k] (bf16) from f32 W.
//   BHEAD=true : W is per-head [N/64, K, 64]  (Wq/Wk/Wv layout)
//   BHEAD=false: W is [K, N]
// 32x32 tiles through LDS; both global sides coalesced.
// =====================================================================
template<bool BHEAD>
__global__ __launch_bounds__(256) void wcvt_kernel(
    const float* __restrict__ W, void* __restrict__ Wtp, int K, int N)
{
    __shared__ __bf16 t[32][33];
    __bf16* Wt = (__bf16*)Wtp;
    const int n0 = blockIdx.x * 32, k0 = blockIdx.y * 32;
    const int tx = threadIdx.x & 31, ty = threadIdx.x >> 5;   // ty 0..7
#pragma unroll
    for (int s = 0; s < 4; ++s) {
        int k = k0 + ty + s * 8;
        int n = n0 + tx;
        size_t src = BHEAD ? ((size_t)(n >> 6) * K + k) * 64 + (n & 63)
                           : (size_t)k * N + n;
        t[ty + s * 8][tx] = (__bf16)W[src];
    }
    __syncthreads();
#pragma unroll
    for (int s = 0; s < 4; ++s) {
        int nl = ty + s * 8;
        Wt[(size_t)(n0 + nl) * K + k0 + tx] = t[tx][nl];
    }
}

// =====================================================================
// GEMM: C[M,N] = act(A[M,K] @ Wt^T + bias),  Wt: bf16 [N][K]
//   A: f32 (converted while staging) or bf16 (async-staged).
//   B staging is always pure async b128 copies (Wt already bf16 [n][k]).
//   Tiles BM=BN=128, BK=32; 256 thr = 8 waves; wave tile 64x32.
//   Double-buffered LDS, one barrier per K-step.
// =====================================================================
template<bool A_BF16, bool RELU, bool OUT_BF16>
__global__ __launch_bounds__(256) void gemm_kernel(
    const void* __restrict__ Ap, const void* __restrict__ Wtp,
    const float* __restrict__ bias, void* __restrict__ Cp,
    int M, int N, int K)
{
    __shared__ __bf16 sA[2][128 * 40];
    __shared__ __bf16 sB[2][128 * 40];   // [n][k]

    const int tid  = threadIdx.x;
    const int lane = tid & 31;
    const int wave = tid >> 5;
    const int wm   = wave >> 2;       // 0..1
    const int wn   = wave & 3;        // 0..3
    const int l    = lane & 15;
    const int hi   = lane >> 4;

    const int m0 = blockIdx.y * 128;
    const int n0 = blockIdx.x * 128;

    auto stage = [&](int buf, int kb) {
        // ---- A tile (128 x 32) ----
        if constexpr (A_BF16) {
            const __bf16* A = (const __bf16*)Ap;
#pragma unroll
            for (int t = 0; t < 2; ++t) {
                int i = tid + t * 256;                 // 512 x 16B
                int r = i >> 2, c8 = (i & 3) * 8;
                async_b128(A + (size_t)(m0 + r) * K + kb + c8,
                           &sA[buf][r * 40 + c8]);
            }
        } else {
            const float* A = (const float*)Ap;
#pragma unroll
            for (int t = 0; t < 4; ++t) {
                int i = tid + t * 256;                 // 1024 x float4
                int r = i >> 3, c4 = (i & 7) * 4;
                float4 f = *(const float4*)(A + (size_t)(m0 + r) * K + kb + c4);
                sA[buf][r * 40 + c4 + 0] = (__bf16)f.x;
                sA[buf][r * 40 + c4 + 1] = (__bf16)f.y;
                sA[buf][r * 40 + c4 + 2] = (__bf16)f.z;
                sA[buf][r * 40 + c4 + 3] = (__bf16)f.w;
            }
        }
        // ---- B tile (128 x 32 of Wt[n][k]): pure async copies ----
        const __bf16* Wt = (const __bf16*)Wtp;
#pragma unroll
        for (int t = 0; t < 2; ++t) {
            int i = tid + t * 256;                     // 512 x 16B
            int n = i >> 2, c8 = (i & 3) * 8;
            async_b128(Wt + (size_t)(n0 + n) * K + kb + c8,
                       &sB[buf][n * 40 + c8]);
        }
    };

    v8f acc[4][2] = {};

    stage(0, 0);
    int buf = 0;
    for (int kb = 0; kb < K; kb += 32) {
        wait_async0();
        __syncthreads();                          // publish buf
        if (kb + 32 < K) stage(buf ^ 1, kb + 32); // overlap next stage w/ compute

        BFrag aF[4], bF[2];
#pragma unroll
        for (int mt = 0; mt < 4; ++mt) {
            int r = wm * 64 + mt * 16 + l;
            aF[mt].h[0] = *(const v8bf*)&sA[buf][r * 40 + hi * 8];
            aF[mt].h[1] = *(const v8bf*)&sA[buf][r * 40 + 16 + hi * 8];
        }
#pragma unroll
        for (int nt = 0; nt < 2; ++nt) {
            int c = wn * 32 + nt * 16 + l;
            bF[nt].h[0] = *(const v8bf*)&sB[buf][c * 40 + hi * 8];
            bF[nt].h[1] = *(const v8bf*)&sB[buf][c * 40 + 16 + hi * 8];
        }
#pragma unroll
        for (int mt = 0; mt < 4; ++mt)
#pragma unroll
            for (int nt = 0; nt < 2; ++nt)
                acc[mt][nt] = __builtin_amdgcn_wmma_f32_16x16x32_bf16(
                    false, aF[mt].v, false, bF[nt].v,
                    (short)0, acc[mt][nt], false, false);
        buf ^= 1;
    }

    // ---- epilogue: bias (+ReLU), store f32 or bf16 ----
#pragma unroll
    for (int mt = 0; mt < 4; ++mt) {
#pragma unroll
        for (int nt = 0; nt < 2; ++nt) {
            int col = n0 + wn * 32 + nt * 16 + l;
            float bv = bias[col];
#pragma unroll
            for (int r = 0; r < 8; ++r) {
                int row = m0 + wm * 64 + mt * 16 + hi * 8 + r;
                float c = acc[mt][nt][r] + bv;
                if (RELU) c = fmaxf(c, 0.0f);
                if (OUT_BF16) ((__bf16*)Cp)[(size_t)row * N + col] = (__bf16)c;
                else          ((float*)Cp)[(size_t)row * N + col]  = c;
            }
        }
    }
}

// =====================================================================
// Flash attention (per batch-head, 128 query rows per block).
//   Q/K/V bf16 [B*S, H*64]. 8 waves x 16 query rows. Double-buffered
//   64-key KV chunks; K async-staged, V transposed via ds stores.
//   Online softmax with 1/8 folded into exp2; row reductions via DPP.
// =====================================================================
__global__ __launch_bounds__(256) void attn_kernel(
    const void* __restrict__ Qp, const void* __restrict__ Kp,
    const void* __restrict__ Vp, void* __restrict__ Op)
{
    const __bf16* Q = (const __bf16*)Qp;
    const __bf16* Kt = (const __bf16*)Kp;
    const __bf16* V = (const __bf16*)Vp;
    __bf16* O = (__bf16*)Op;

    __shared__ __bf16 sK[2][64 * 72];     // [kv][d]
    __shared__ __bf16 sV[2][64 * 72];     // [d][kv]
    __shared__ __bf16 sP[8 * 16 * 72];    // per-wave 16x64 P tile

    const int tid  = threadIdx.x;
    const int lane = tid & 31;
    const int wave = tid >> 5;
    const int l    = lane & 15;
    const int hi   = lane >> 4;

    const int b  = blockIdx.z, h = blockIdx.y;
    const int q0 = blockIdx.x * 128;
    const int qrow = q0 + wave * 16 + l;
    const size_t qoff = (size_t)(b * 1024 + qrow) * 1024 + h * 64;

    const float SC = 0.125f * LOG2E;      // scale folded into exp2 args

    auto stageKV = [&](int buf, int kv0) {
#pragma unroll
        for (int t = 0; t < 2; ++t) {
            int i = tid + t * 256;
            int r = i >> 3, c8 = (i & 7) * 8;
            async_b128(Kt + (size_t)(b * 1024 + kv0 + r) * 1024 + h * 64 + c8,
                       &sK[buf][r * 72 + c8]);
        }
#pragma unroll
        for (int t = 0; t < 4; ++t) {
            int i = tid + t * 256;
            int r = i >> 4, d4 = (i & 15) * 4;
            union { uint2 u; __bf16 e[4]; } vv;
            vv.u = *(const uint2*)(V + (size_t)(b * 1024 + kv0 + r) * 1024 + h * 64 + d4);
            sV[buf][(d4 + 0) * 72 + r] = vv.e[0];
            sV[buf][(d4 + 1) * 72 + r] = vv.e[1];
            sV[buf][(d4 + 2) * 72 + r] = vv.e[2];
            sV[buf][(d4 + 3) * 72 + r] = vv.e[3];
        }
    };

    BFrag qF[2];
#pragma unroll
    for (int d0 = 0; d0 < 2; ++d0) {
        qF[d0].h[0] = *(const v8bf*)(Q + qoff + d0 * 32 + hi * 8);
        qF[d0].h[1] = *(const v8bf*)(Q + qoff + d0 * 32 + 16 + hi * 8);
    }

    float m[8], lsum[8];
    v8f o[4] = {};
#pragma unroll
    for (int j = 0; j < 8; ++j) { m[j] = -1e30f; lsum[j] = 0.0f; }

    __bf16* myP = sP + wave * 16 * 72;

    stageKV(0, 0);
    int buf = 0;
    for (int kv0 = 0; kv0 < 1024; kv0 += 64) {
        wait_async0();
        __syncthreads();                              // publish buf
        if (kv0 + 64 < 1024) stageKV(buf ^ 1, kv0 + 64);

        // ---- S = Q K^T (16x64 per wave, raw scores) ----
        v8f s[4] = {};
#pragma unroll
        for (int nt = 0; nt < 4; ++nt) {
            int c = nt * 16 + l;
#pragma unroll
            for (int d0 = 0; d0 < 2; ++d0) {
                BFrag kF;
                kF.h[0] = *(const v8bf*)&sK[buf][c * 72 + d0 * 32 + hi * 8];
                kF.h[1] = *(const v8bf*)&sK[buf][c * 72 + d0 * 32 + 16 + hi * 8];
                s[nt] = __builtin_amdgcn_wmma_f32_16x16x32_bf16(
                    false, qF[d0].v, false, kF.v, (short)0, s[nt], false, false);
            }
        }

        // ---- online softmax (row = j + 8*hi, cols across 16 lanes) ----
        float fac[8];
#pragma unroll
        for (int j = 0; j < 8; ++j) {
            float rm = fmaxf(fmaxf(s[0][j], s[1][j]), fmaxf(s[2][j], s[3][j]));
            rm = dpp_max16(rm);
            float mn = fmaxf(m[j], rm);
            fac[j] = exp2f((m[j] - mn) * SC);
            m[j] = mn;
        }
        float rs[8];
#pragma unroll
        for (int j = 0; j < 8; ++j) rs[j] = 0.0f;
#pragma unroll
        for (int nt = 0; nt < 4; ++nt) {
#pragma unroll
            for (int j = 0; j < 8; ++j) {
                float p = exp2f((s[nt][j] - m[j]) * SC);
                rs[j] += p;
                myP[(hi * 8 + j) * 72 + nt * 16 + l] = (__bf16)p;
            }
        }
#pragma unroll
        for (int j = 0; j < 8; ++j) {
            float r = dpp_sum16(rs[j]);
            lsum[j] = lsum[j] * fac[j] + r;
#pragma unroll
            for (int nt = 0; nt < 4; ++nt) o[nt][j] *= fac[j];
        }

        // ---- O += P V ----
#pragma unroll
        for (int kvp = 0; kvp < 2; ++kvp) {
            BFrag pF;
            pF.h[0] = *(const v8bf*)&myP[l * 72 + kvp * 32 + hi * 8];
            pF.h[1] = *(const v8bf*)&myP[l * 72 + kvp * 32 + 16 + hi * 8];
#pragma unroll
            for (int nt = 0; nt < 4; ++nt) {
                int c = nt * 16 + l;
                BFrag vF;
                vF.h[0] = *(const v8bf*)&sV[buf][c * 72 + kvp * 32 + hi * 8];
                vF.h[1] = *(const v8bf*)&sV[buf][c * 72 + kvp * 32 + 16 + hi * 8];
                o[nt] = __builtin_amdgcn_wmma_f32_16x16x32_bf16(
                    false, pF.v, false, vF.v, (short)0, o[nt], false, false);
            }
        }
        buf ^= 1;
    }

    // ---- normalize, store bf16 head-concat context ----
#pragma unroll
    for (int nt = 0; nt < 4; ++nt) {
#pragma unroll
        for (int j = 0; j < 8; ++j) {
            int row = q0 + wave * 16 + hi * 8 + j;
            float val = o[nt][j] / lsum[j];
            O[(size_t)(b * 1024 + row) * 1024 + h * 64 + nt * 16 + l] = (__bf16)val;
        }
    }
}

// =====================================================================
// y = LayerNorm(x + r) over D=1024, one 256-thread block per row.
// DPP for in-row reduction, one shuffle for the cross-half step.
// =====================================================================
__global__ __launch_bounds__(256) void add_ln_kernel(
    const float* __restrict__ X, const float* __restrict__ R,
    const float* __restrict__ G, const float* __restrict__ Bt,
    float* __restrict__ Y)
{
    __shared__ float red[8];
    const int row = blockIdx.x;
    const int tid = threadIdx.x;
    const size_t base = (size_t)row * 1024 + tid * 4;
    float4 x4 = *(const float4*)&X[base];
    float4 r4 = *(const float4*)&R[base];
    float v0 = x4.x + r4.x, v1 = x4.y + r4.y, v2 = x4.z + r4.z, v3 = x4.w + r4.w;

    float s = v0 + v1 + v2 + v3;
    s = dpp_sum16(s);
    s += __shfl_xor(s, 16, 32);
    if ((tid & 31) == 0) red[tid >> 5] = s;
    __syncthreads();
    float tot = 0.0f;
#pragma unroll
    for (int i = 0; i < 8; ++i) tot += red[i];
    float mean = tot * (1.0f / 1024.0f);
    __syncthreads();

    float d0 = v0 - mean, d1 = v1 - mean, d2 = v2 - mean, d3 = v3 - mean;
    float q = d0 * d0 + d1 * d1 + d2 * d2 + d3 * d3;
    q = dpp_sum16(q);
    q += __shfl_xor(q, 16, 32);
    if ((tid & 31) == 0) red[tid >> 5] = q;
    __syncthreads();
    float vt = 0.0f;
#pragma unroll
    for (int i = 0; i < 8; ++i) vt += red[i];
    float rstd = rsqrtf(vt * (1.0f / 1024.0f) + 1e-5f);

    float4 g4 = *(const float4*)&G[tid * 4];
    float4 b4 = *(const float4*)&Bt[tid * 4];
    float4 out;
    out.x = d0 * rstd * g4.x + b4.x;
    out.y = d1 * rstd * g4.y + b4.y;
    out.z = d2 * rstd * g4.z + b4.z;
    out.w = d3 * rstd * g4.w + b4.w;
    *(float4*)&Y[base] = out;
}

// =====================================================================
extern "C" void kernel_launch(void* const* d_in, const int* in_sizes, int n_in,
                              void* d_out, int out_size, void* d_ws, size_t ws_size,
                              hipStream_t stream)
{
    const float* q   = (const float*)d_in[0];
    const float* mA  = (const float*)d_in[1];
    const float* mB  = (const float*)d_in[2];
    const float* Wq1 = (const float*)d_in[3];  const float* bq1 = (const float*)d_in[4];
    const float* Wk1 = (const float*)d_in[5];  const float* bk1 = (const float*)d_in[6];
    const float* Wv1 = (const float*)d_in[7];  const float* bv1 = (const float*)d_in[8];
    const float* Wo1 = (const float*)d_in[9];  const float* bo1 = (const float*)d_in[10];
    const float* Wq2 = (const float*)d_in[11]; const float* bq2 = (const float*)d_in[12];
    const float* Wk2 = (const float*)d_in[13]; const float* bk2 = (const float*)d_in[14];
    const float* Wv2 = (const float*)d_in[15]; const float* bv2 = (const float*)d_in[16];
    const float* Wo2 = (const float*)d_in[17]; const float* bo2 = (const float*)d_in[18];
    const float* W1  = (const float*)d_in[19]; const float* b1  = (const float*)d_in[20];
    const float* W2  = (const float*)d_in[21]; const float* b2  = (const float*)d_in[22];
    const float* g1  = (const float*)d_in[23]; const float* be1 = (const float*)d_in[24];
    const float* g2  = (const float*)d_in[25]; const float* be2 = (const float*)d_in[26];
    const float* g3  = (const float*)d_in[27]; const float* be3 = (const float*)d_in[28];
    (void)in_sizes; (void)n_in; (void)out_size; (void)ws_size;

    char* ws = (char*)d_ws;
    size_t off = 0;
    auto carve = [&](size_t bytes) {
        void* p = ws + off;
        off += (bytes + 255) & ~(size_t)255;
        return p;
    };
    const size_t ACT_BF = (size_t)4096 * 1024 * 2;
    const size_t ACT_F  = (size_t)4096 * 1024 * 4;
    const size_t HID_BF = (size_t)4096 * 4096 * 2;
    const size_t W_SQ   = (size_t)1024 * 1024 * 2;   // 1024x1024 bf16
    const size_t W_BIG  = (size_t)1024 * 4096 * 2;   // 1024x4096 bf16

    // bf16 transposed weights Wt[n][k]
    void* tWq1 = carve(W_SQ); void* tWk1 = carve(W_SQ); void* tWv1 = carve(W_SQ);
    void* tWo1 = carve(W_SQ);
    void* tWq2 = carve(W_SQ); void* tWk2 = carve(W_SQ); void* tWv2 = carve(W_SQ);
    void* tWo2 = carve(W_SQ);
    void* tW1  = carve(W_BIG);
    void* tW2  = carve(W_BIG);

    void* Qb = carve(ACT_BF);
    void* Kb = carve(ACT_BF);
    void* Vb = carve(ACT_BF);
    void* Cx = carve(ACT_BF);   // attention context (bf16, head-concat)
    void* Sf = carve(ACT_F);    // f32 scratch (proj / ffn2 outputs)
    void* X1 = carve(ACT_F);
    void* X2 = carve(ACT_F);
    void* Hb = carve(HID_BF);   // FFN hidden (bf16)

    dim3 blk(256);
    dim3 gWsq(1024 / 32, 1024 / 32);   // weight cvt, K=1024 N=1024
    dim3 gW1 (4096 / 32, 1024 / 32);   // W1: K=1024 N=4096
    dim3 gW2 (1024 / 32, 4096 / 32);   // W2: K=4096 N=1024
    dim3 gP(1024 / 128, 4096 / 128);   // N=1024 GEMMs
    dim3 gF1(4096 / 128, 4096 / 128);  // FFN1 (N=4096)
    dim3 gA(1024 / 128, 16, 4);        // attention

    // ---- one-time weight transpose+convert to bf16 [n][k] ----
    wcvt_kernel<true ><<<gWsq, blk, 0, stream>>>(Wq1, tWq1, 1024, 1024);
    wcvt_kernel<true ><<<gWsq, blk, 0, stream>>>(Wk1, tWk1, 1024, 1024);
    wcvt_kernel<true ><<<gWsq, blk, 0, stream>>>(Wv1, tWv1, 1024, 1024);
    wcvt_kernel<false><<<gWsq, blk, 0, stream>>>(Wo1, tWo1, 1024, 1024);
    wcvt_kernel<true ><<<gWsq, blk, 0, stream>>>(Wq2, tWq2, 1024, 1024);
    wcvt_kernel<true ><<<gWsq, blk, 0, stream>>>(Wk2, tWk2, 1024, 1024);
    wcvt_kernel<true ><<<gWsq, blk, 0, stream>>>(Wv2, tWv2, 1024, 1024);
    wcvt_kernel<false><<<gWsq, blk, 0, stream>>>(Wo2, tWo2, 1024, 1024);
    wcvt_kernel<false><<<gW1,  blk, 0, stream>>>(W1,  tW1,  1024, 4096);
    wcvt_kernel<false><<<gW2,  blk, 0, stream>>>(W2,  tW2,  4096, 1024);

    // ---- MHA 1: Q from query_modality, K/V from modality_A ----
    gemm_kernel<false, false, true ><<<gP, blk, 0, stream>>>(q,  tWq1, bq1, Qb, 4096, 1024, 1024);
    gemm_kernel<false, false, true ><<<gP, blk, 0, stream>>>(mA, tWk1, bk1, Kb, 4096, 1024, 1024);
    gemm_kernel<false, false, true ><<<gP, blk, 0, stream>>>(mA, tWv1, bv1, Vb, 4096, 1024, 1024);
    attn_kernel<<<gA, blk, 0, stream>>>(Qb, Kb, Vb, Cx);
    gemm_kernel<true,  false, false><<<gP, blk, 0, stream>>>(Cx, tWo1, bo1, Sf, 4096, 1024, 1024);
    add_ln_kernel<<<4096, blk, 0, stream>>>(q, (const float*)Sf, g1, be1, (float*)X1);

    // ---- MHA 2: Q from x1, K/V from modality_B ----
    gemm_kernel<false, false, true ><<<gP, blk, 0, stream>>>(X1, tWq2, bq2, Qb, 4096, 1024, 1024);
    gemm_kernel<false, false, true ><<<gP, blk, 0, stream>>>(mB, tWk2, bk2, Kb, 4096, 1024, 1024);
    gemm_kernel<false, false, true ><<<gP, blk, 0, stream>>>(mB, tWv2, bv2, Vb, 4096, 1024, 1024);
    attn_kernel<<<gA, blk, 0, stream>>>(Qb, Kb, Vb, Cx);
    gemm_kernel<true,  false, false><<<gP, blk, 0, stream>>>(Cx, tWo2, bo2, Sf, 4096, 1024, 1024);
    add_ln_kernel<<<4096, blk, 0, stream>>>((const float*)X1, (const float*)Sf, g2, be2, (float*)X2);

    // ---- FFN ----
    gemm_kernel<false, true,  true ><<<gF1, blk, 0, stream>>>(X2, tW1, b1, Hb, 4096, 4096, 1024);
    gemm_kernel<true,  false, false><<<gP, blk, 0, stream>>>(Hb, tW2, b2, Sf, 4096, 1024, 4096);
    add_ln_kernel<<<4096, blk, 0, stream>>>((const float*)X2, (const float*)Sf, g3, be3, (float*)d_out);
}
